// GCN_4947802325635
// MI455X (gfx1250) — matrix operand order repaired and checked by reading, not compiled, not run on hardware
//
#include <hip/hip_runtime.h>
#include <hip/hip_bf16.h>
#include <stdint.h>

typedef float  v8f   __attribute__((ext_vector_type(8)));
typedef __bf16 v16bf __attribute__((ext_vector_type(16)));

constexpr int N_NODES  = 100000;
constexpr int N_EDGES  = 1600000;
constexpr int N_GRAPHS = 2000;
constexpr int D_IN     = 600;
constexpr int D_H      = 300;
constexpr int NPAD     = 320;   // hidden width padded to 20 N-tiles (zeros beyond 300)
constexpr int K1PAD    = 608;   // 19 k-tiles of 32 for layer 1
constexpr int K2PAD    = 320;   // 10 k-tiles of 32 for layer 2

__device__ __forceinline__ unsigned short f2bf(float f) {
    unsigned u = __float_as_uint(f);
    u += 0x7FFFu + ((u >> 16) & 1u);   // round-to-nearest-even
    return (unsigned short)(u >> 16);
}

// ---- degree accumulation (atomic) ----
__global__ void deg_kernel(const long long* __restrict__ src,
                           const long long* __restrict__ dst,
                           float* __restrict__ degO, float* __restrict__ degI, int nE) {
    int e = blockIdx.x * blockDim.x + threadIdx.x;
    if (e < nE) {
        atomicAdd(&degO[(int)src[e]], 1.0f);
        atomicAdd(&degI[(int)dst[e]], 1.0f);
    }
}

__global__ void invsqrt_kernel(float* __restrict__ d, int n) {
    int i = blockIdx.x * blockDim.x + threadIdx.x;
    if (i < n) { float v = d[i]; d[i] = (v > 0.f) ? rsqrtf(v) : 0.f; }
}

// ---- weight prep: W[K][N] fp32 -> Wt[NPAD][Kpad] bf16 (zero padded) ----
__global__ void prep_wt(const float* __restrict__ W, unsigned short* __restrict__ Wt,
                        int K, int N, int Kpad) {
    int i = blockIdx.x * blockDim.x + threadIdx.x;
    if (i >= NPAD * Kpad) return;
    int n = i / Kpad, k = i % Kpad;
    float v = (n < N && k < K) ? W[k * N + n] : 0.f;
    Wt[i] = f2bf(v);
}

// ---- WMMA bf16 GEMM, double-buffered LDS software pipeline ----
// C[M][NPAD] = (A[M][lda] fp32 -> bf16) x Bt^T, per-row scale fused in epilogue.
// block: 256 threads = 8 waves in 4(M) x 2(N) grid; block tile 128x64; wave tile 32x32.
__global__ __launch_bounds__(256)
void gemm_bf16(const float* __restrict__ A, int lda, int klim,
               const unsigned short* __restrict__ Bt, int kpad, int ktiles,
               const float* __restrict__ scale, float* __restrict__ C) {
    __shared__ unsigned short As[2][128 * 48];  // 32 cols used, stride 48 (96B, 16B aligned)
    __shared__ unsigned short Bs[2][64 * 48];

    const int tid   = threadIdx.x;
    const int mBase = blockIdx.y * 128;
    const int nBase = blockIdx.x * 64;
    const int wave  = tid >> 5, lane = tid & 31;
    const int wm = wave & 3, wn = wave >> 2;
    const int lm = lane & 15;
    const int hi = lane >> 4;                // 0: lanes 0-15, 1: lanes 16-31

    // per-thread staging coordinates (fixed across k-tiles)
    const int bnl = tid >> 2, bk8 = (tid & 3) * 8;

    auto loadTile = [&](int k0, float4* va, uint4& vb) {
        #pragma unroll
        for (int t = 0; t < 4; ++t) {
            int s = tid + t * 256;               // 0..1023 float4 slots of the 128x32 A tile
            int r = s >> 3, c4 = (s & 7) * 4;
            int grow = mBase + r, gcol = k0 + c4;
            va[t] = make_float4(0.f, 0.f, 0.f, 0.f);
            if (grow < N_NODES && gcol < klim)
                va[t] = *reinterpret_cast<const float4*>(A + (size_t)grow * lda + gcol);
        }
        vb = *reinterpret_cast<const uint4*>(Bt + (size_t)(nBase + bnl) * kpad + k0 + bk8);
    };
    auto storeTile = [&](int buf, const float4* va, const uint4& vb) {
        #pragma unroll
        for (int t = 0; t < 4; ++t) {
            int s = tid + t * 256;
            int r = s >> 3, c4 = (s & 7) * 4;
            uint2 p;
            p.x = (unsigned)f2bf(va[t].x) | ((unsigned)f2bf(va[t].y) << 16);
            p.y = (unsigned)f2bf(va[t].z) | ((unsigned)f2bf(va[t].w) << 16);
            *reinterpret_cast<uint2*>(&As[buf][r * 48 + c4]) = p;
        }
        *reinterpret_cast<uint4*>(&Bs[buf][bnl * 48 + bk8]) = vb;
    };

    v8f acc[2][2] = {};
    float4 va[4];
    uint4  vb;

    // prologue: stage k-tile 0
    loadTile(0, va, vb);
    storeTile(0, va, vb);
    __syncthreads();

    for (int kt = 0; kt < ktiles; ++kt) {
        const int  cur  = kt & 1;
        const bool more = (kt + 1) < ktiles;

        // issue global loads for next tile early (overlap with WMMA on current tile)
        if (more) loadTile((kt + 1) * 32, va, vb);

        // fragments per ISA 7.12.2:
        // A: lane<16 -> K[0:8)+K[16:24) of row M=lm; lane>=16 -> K[8:16)+K[24:32)
        // B: lane holds column N=lm, K[0:8)+K[16:24) (lanes 0-15) or K[8:16)+K[24:32) (lanes 16-31)
        union Frag { v16bf v; uint4 q[2]; } aF[2], bF[2];
        #pragma unroll
        for (int i = 0; i < 2; ++i) {
            const unsigned short* p = &As[cur][(wm * 32 + i * 16 + lm) * 48 + hi * 8];
            aF[i].q[0] = *reinterpret_cast<const uint4*>(p);
            aF[i].q[1] = *reinterpret_cast<const uint4*>(p + 16);
        }
        #pragma unroll
        for (int j = 0; j < 2; ++j) {
            const unsigned short* p = &Bs[cur][(wn * 32 + j * 16 + lm) * 48 + hi * 8];
            bF[j].q[0] = *reinterpret_cast<const uint4*>(p);
            bF[j].q[1] = *reinterpret_cast<const uint4*>(p + 16);
        }

        #pragma unroll
        for (int i = 0; i < 2; ++i)
            #pragma unroll
            for (int j = 0; j < 2; ++j)
                acc[i][j] = __builtin_amdgcn_wmma_f32_16x16x32_bf16(
                    false, aF[i].v, false, bF[j].v, (short)0, acc[i][j], false, false);

        if (more) {
            storeTile(1 - cur, va, vb);
            __syncthreads();
        }
    }

    // epilogue: C/D layout -> VGPR r holds M = r + 8*hi, N = lm; fuse src-degree scale
    #pragma unroll
    for (int i = 0; i < 2; ++i) {
        int rbase = mBase + wm * 32 + i * 16 + hi * 8;
        #pragma unroll
        for (int r = 0; r < 8; ++r) {
            int grow = rbase + r;
            if (grow < N_NODES) {
                float s = scale[grow];
                int col = nBase + wn * 32 + lm;
                C[(size_t)grow * NPAD + col]      = acc[i][0][r] * s;
                C[(size_t)grow * NPAD + col + 16] = acc[i][1][r] * s;
            }
        }
    }
}

// ---- edge scatter: AGG[dst] += H[src], one wave per edge ----
__global__ __launch_bounds__(256)
void scatter_kernel(const float* __restrict__ H, float* __restrict__ AGG,
                    const long long* __restrict__ src, const long long* __restrict__ dst, int nE) {
    int e = blockIdx.x * 8 + (threadIdx.x >> 5);
    if (e >= nE) return;
    int lane = threadIdx.x & 31;
    const float* hs = H  + (size_t)src[e] * NPAD;
    float*       ad = AGG + (size_t)dst[e] * NPAD;
    for (int c = lane; c < D_H; c += 32)
        atomicAdd(&ad[c], hs[c]);
}

// ---- out[i,:] = relu(AGG[i,:]*inv_in[i] + b) (pad cols forced to 0) ----
__global__ void relu_bias_kernel(const float* __restrict__ AGG, const float* __restrict__ invin,
                                 const float* __restrict__ b, float* __restrict__ out) {
    int i = blockIdx.x * blockDim.x + threadIdx.x;   // over N_NODES * (NPAD/4) float4s
    if (i >= N_NODES * (NPAD / 4)) return;
    int node = i / (NPAD / 4);
    int c4 = (i % (NPAD / 4)) * 4;
    float4 o = make_float4(0.f, 0.f, 0.f, 0.f);
    if (c4 < D_H) {
        float s = invin[node];
        float4 a  = *reinterpret_cast<const float4*>(AGG + (size_t)node * NPAD + c4);
        float4 bb = *reinterpret_cast<const float4*>(b + c4);
        o.x = fmaxf(a.x * s + bb.x, 0.f);
        o.y = fmaxf(a.y * s + bb.y, 0.f);
        o.z = fmaxf(a.z * s + bb.z, 0.f);
        o.w = fmaxf(a.w * s + bb.w, 0.f);
    }
    *reinterpret_cast<float4*>(out + (size_t)node * NPAD + c4) = o;
}

// ---- 300->2 head + softmax + atomic graph pooling ----
__global__ __launch_bounds__(256)
void head_kernel(const float* __restrict__ H, const float* __restrict__ Wfc,
                 const float* __restrict__ bfc, const long long* __restrict__ gid,
                 float* __restrict__ pooled, float* __restrict__ cnt) {
    __shared__ float wf[D_H * 2];
    for (int i = threadIdx.x; i < D_H * 2; i += blockDim.x) wf[i] = Wfc[i];
    __syncthreads();
    int n = blockIdx.x * blockDim.x + threadIdx.x;
    if (n >= N_NODES) return;
    const float* h = H + (size_t)n * NPAD;
    float l0 = bfc[0], l1 = bfc[1];
    for (int k = 0; k < D_H; ++k) {
        float hv = h[k];
        l0 += hv * wf[2 * k];
        l1 += hv * wf[2 * k + 1];
    }
    float m  = fmaxf(l0, l1);
    float e0 = expf(l0 - m), e1 = expf(l1 - m);
    float inv = 1.f / (e0 + e1);
    int g = (int)gid[n];
    atomicAdd(&pooled[2 * g],     e0 * inv);
    atomicAdd(&pooled[2 * g + 1], e1 * inv);
    atomicAdd(&cnt[g], 1.f);
}

__global__ void finalize_kernel(const float* __restrict__ pooled, const float* __restrict__ cnt,
                                float* __restrict__ out) {
    int g = blockIdx.x * blockDim.x + threadIdx.x;
    if (g < N_GRAPHS) {
        float c = fmaxf(cnt[g], 1.f);
        out[2 * g]     = pooled[2 * g] / c;
        out[2 * g + 1] = pooled[2 * g + 1] / c;
    }
}

extern "C" void kernel_launch(void* const* d_in, const int* in_sizes, int n_in,
                              void* d_out, int out_size, void* d_ws, size_t ws_size,
                              hipStream_t stream) {
    const float*     X   = (const float*)d_in[0];
    const float*     W1  = (const float*)d_in[1];
    const float*     b1  = (const float*)d_in[2];
    const float*     W2  = (const float*)d_in[3];
    const float*     b2  = (const float*)d_in[4];
    const float*     Wfc = (const float*)d_in[5];
    const float*     bfc = (const float*)d_in[6];
    const long long* src = (const long long*)d_in[7];
    const long long* dst = (const long long*)d_in[8];
    const long long* gid = (const long long*)d_in[9];
    (void)in_sizes; (void)n_in; (void)out_size; (void)ws_size;

    char* ws = (char*)d_ws;
    size_t off = 0;
    auto salloc = [&](size_t bytes) -> void* {
        void* p = ws + off;
        off = (off + bytes + 255) & ~(size_t)255;
        return p;
    };
    float*          invO   = (float*)salloc((size_t)N_NODES * 4);
    float*          invI   = (float*)salloc((size_t)N_NODES * 4);
    unsigned short* W1T    = (unsigned short*)salloc((size_t)NPAD * K1PAD * 2);
    unsigned short* W2T    = (unsigned short*)salloc((size_t)NPAD * K2PAD * 2);
    float*          bufA   = (float*)salloc((size_t)N_NODES * NPAD * 4);
    float*          bufB   = (float*)salloc((size_t)N_NODES * NPAD * 4);
    float*          pooled = (float*)salloc((size_t)N_GRAPHS * 2 * 4);
    float*          cnt    = (float*)salloc((size_t)N_GRAPHS * 4);

    // degrees -> inverse sqrt norms
    hipMemsetAsync(invO, 0, (size_t)N_NODES * 4, stream);
    hipMemsetAsync(invI, 0, (size_t)N_NODES * 4, stream);
    deg_kernel<<<(N_EDGES + 255) / 256, 256, 0, stream>>>(src, dst, invO, invI, N_EDGES);
    invsqrt_kernel<<<(N_NODES + 255) / 256, 256, 0, stream>>>(invO, N_NODES);
    invsqrt_kernel<<<(N_NODES + 255) / 256, 256, 0, stream>>>(invI, N_NODES);

    // bf16 transposed, zero-padded weights
    prep_wt<<<((NPAD * K1PAD) + 255) / 256, 256, 0, stream>>>(W1, W1T, D_IN, D_H, K1PAD);
    prep_wt<<<((NPAD * K2PAD) + 255) / 256, 256, 0, stream>>>(W2, W2T, D_H, D_H, K2PAD);

    dim3 gg(NPAD / 64, (N_NODES + 127) / 128);

    // layer 1
    gemm_bf16<<<gg, 256, 0, stream>>>(X, D_IN, D_IN, W1T, K1PAD, K1PAD / 32, invO, bufA);
    hipMemsetAsync(bufB, 0, (size_t)N_NODES * NPAD * 4, stream);
    scatter_kernel<<<(N_EDGES + 7) / 8, 256, 0, stream>>>(bufA, bufB, src, dst, N_EDGES);
    relu_bias_kernel<<<((N_NODES * (NPAD / 4)) + 255) / 256, 256, 0, stream>>>(bufB, invI, b1, bufA);

    // layer 2
    gemm_bf16<<<gg, 256, 0, stream>>>(bufA, NPAD, NPAD, W2T, K2PAD, K2PAD / 32, invO, bufB);
    hipMemsetAsync(bufA, 0, (size_t)N_NODES * NPAD * 4, stream);
    scatter_kernel<<<(N_EDGES + 7) / 8, 256, 0, stream>>>(bufB, bufA, src, dst, N_EDGES);
    relu_bias_kernel<<<((N_NODES * (NPAD / 4)) + 255) / 256, 256, 0, stream>>>(bufA, invI, b2, bufB);

    // head + pooling
    hipMemsetAsync(pooled, 0, (size_t)N_GRAPHS * 2 * 4, stream);
    hipMemsetAsync(cnt, 0, (size_t)N_GRAPHS * 4, stream);
    head_kernel<<<(N_NODES + 255) / 256, 256, 0, stream>>>(bufB, Wfc, bfc, gid, pooled, cnt);
    finalize_kernel<<<(N_GRAPHS + 255) / 256, 256, 0, stream>>>(pooled, cnt, (float*)d_out);
}
